// Attention_3874060501749
// MI455X (gfx1250) — matrix operand order repaired and checked by reading, not compiled, not run on hardware
//
#include <hip/hip_runtime.h>

// ---------------------------------------------------------------------------
// ViT attention block for MI455X (gfx1250, wave32, WMMA).
//   K1: qkv = x @ qkv_w^T + b  -> Q(scaled),K row-major, V transposed, bf16
//   K2: flash attention per (batch, head), 32 q-rows per wave (2 q-tiles)
//   K3: out = attn @ proj_w^T + proj_b  (fp32 out)
// All matrix math: v_wmma_f32_16x16x32_bf16 (f32 accumulate).
// GEMMs use 64x64 per-wave register tiles: 16 WMMA per 8 fragment loads.
// ---------------------------------------------------------------------------

typedef __attribute__((ext_vector_type(16))) __bf16 v16bf;
typedef __attribute__((ext_vector_type(8)))  __bf16 bf16x8;
typedef __attribute__((ext_vector_type(8)))  float  v8f;
typedef __attribute__((ext_vector_type(4)))  float  f32x4;

#define DIM   768
#define NH    12
#define HD    64
#define SEQ   4096
#define BATCH 2
#define QKV_ELEMS (BATCH * NH * SEQ * HD)   // 6,291,456 bf16 elements per tensor

static __device__ __forceinline__ int lane_id() { return threadIdx.x & 31; }

static __device__ __forceinline__ v8f vzero() {
  v8f z;
#pragma unroll
  for (int i = 0; i < 8; ++i) z[i] = 0.0f;
  return z;
}

static __device__ __forceinline__ v8f wmma_bf16(v16bf a, v16bf b, v8f c) {
  // (neg_a, A, neg_b, B, c_mod, C, reuse_a, reuse_b)
  return __builtin_amdgcn_wmma_f32_16x16x32_bf16(false, a, false, b,
                                                 (short)0, c, false, false);
}

// A-fragment 16x32 (ISA 7.12.2: lanes 0-15 row m, K = 8*hi + j / 16+8*hi+(j-8))
static __device__ __forceinline__ v16bf frag_a_f32(const float* M, int ld, int r0, int k0) {
  const int l = lane_id();
  const float* p = M + (size_t)(r0 + (l & 15)) * ld + k0;
  const int hb = (l >> 4) * 8;
  f32x4 x0 = *(const f32x4*)(p + hb);
  f32x4 x1 = *(const f32x4*)(p + hb + 4);
  f32x4 x2 = *(const f32x4*)(p + 16 + hb);
  f32x4 x3 = *(const f32x4*)(p + 16 + hb + 4);
  v16bf o;
#pragma unroll
  for (int i = 0; i < 4; ++i) {
    o[i]      = (__bf16)x0[i];
    o[4 + i]  = (__bf16)x1[i];
    o[8 + i]  = (__bf16)x2[i];
    o[12 + i] = (__bf16)x3[i];
  }
  return o;
}

static __device__ __forceinline__ v16bf frag_a_bf16(const __bf16* M, int ld, int r0, int k0) {
  const int l = lane_id();
  const __bf16* p = M + (size_t)(r0 + (l & 15)) * ld + k0;
  const int hb = (l >> 4) * 8;
  bf16x8 x0 = *(const bf16x8*)(p + hb);
  bf16x8 x1 = *(const bf16x8*)(p + 16 + hb);
  v16bf o;
#pragma unroll
  for (int i = 0; i < 8; ++i) { o[i] = x0[i]; o[8 + i] = x1[i]; }
  return o;
}

// B-fragment 32x16: lane holds column n = lane&15; K = 8*hi + j / 16+8*hi+(j-8)
static __device__ __forceinline__ v16bf frag_b_f32(const float* M, int ld, int n0, int k0) {
  const int l = lane_id();
  const float* p = M + (size_t)(n0 + (l & 15)) * ld + k0 + (l >> 4) * 8;
  v16bf o;
#pragma unroll
  for (int i = 0; i < 16; i += 4) {
    f32x4 x = *(const f32x4*)(p + i + (i >> 3) * 8);
    o[i] = (__bf16)x[0]; o[i+1] = (__bf16)x[1]; o[i+2] = (__bf16)x[2]; o[i+3] = (__bf16)x[3];
  }
  return o;
}

static __device__ __forceinline__ v16bf frag_b_bf16(const __bf16* M, int ld, int n0, int k0) {
  const int l = lane_id();
  const __bf16* p = M + (size_t)(n0 + (l & 15)) * ld + k0 + (l >> 4) * 8;
  bf16x8 x0 = *(const bf16x8*)(p);
  bf16x8 x1 = *(const bf16x8*)(p + 16);
  v16bf o;
#pragma unroll
  for (int i = 0; i < 8; ++i) { o[i] = x0[i]; o[8 + i] = x1[i]; }
  return o;
}

// ---------------------------------------------------------------------------
// Kernel 1: QKV GEMM. grid=(8192/128, 2304/256), block=256 (8 waves, 2x4).
// Each wave: 64x64 output tile = 4x4 WMMA tiles; K-loop 768/32.
// ---------------------------------------------------------------------------
__global__ __launch_bounds__(256) void qkv_kernel(
    const float* __restrict__ x, const float* __restrict__ w,
    const float* __restrict__ bias,
    __bf16* __restrict__ q, __bf16* __restrict__ kk, __bf16* __restrict__ vt) {
  const int wv = threadIdx.x >> 5;
  const int m0 = blockIdx.x * 128 + (wv >> 2) * 64;
  const int n0 = blockIdx.y * 256 + (wv & 3) * 64;
  const int l = lane_id();
  const int cn = l & 15, hi = l >> 4;

  v8f acc[4][4];
#pragma unroll
  for (int i = 0; i < 4; ++i)
#pragma unroll
    for (int j = 0; j < 4; ++j) acc[i][j] = vzero();

  for (int k0 = 0; k0 < DIM; k0 += 32) {
    if (k0 + 32 < DIM) {
      __builtin_prefetch(x + (size_t)(m0 + cn) * DIM + k0 + 32, 0, 1);
      __builtin_prefetch(w + (size_t)(n0 + cn) * DIM + k0 + 32, 0, 1);
    }
    v16bf a[4], b[4];
#pragma unroll
    for (int i = 0; i < 4; ++i) a[i] = frag_a_f32(x, DIM, m0 + 16 * i, k0);
#pragma unroll
    for (int j = 0; j < 4; ++j) b[j] = frag_b_f32(w, DIM, n0 + 16 * j, k0);
#pragma unroll
    for (int i = 0; i < 4; ++i)
#pragma unroll
      for (int j = 0; j < 4; ++j) acc[i][j] = wmma_bf16(a[i], b[j], acc[i][j]);
  }

#pragma unroll
  for (int j = 0; j < 4; ++j) {
    const int d = n0 + j * 16 + cn;
    const float bv = bias[d];
    const int which = d / DIM;          // 0=q, 1=k, 2=v (uniform within tile)
    const int c = d % DIM;
    const int head = c >> 6, hc = c & 63;
    const float scale = (which == 0) ? 0.125f : 1.0f;  // hd^-0.5 folded into Q
#pragma unroll
    for (int i = 0; i < 4; ++i) {
#pragma unroll
      for (int r = 0; r < 8; ++r) {
        const int m = m0 + i * 16 + hi * 8 + r;
        const int bb = m >> 12, tok = m & (SEQ - 1);
        const __bf16 v = (__bf16)((acc[i][j][r] + bv) * scale);
        const size_t idx_qk = ((size_t)(bb * NH + head) * SEQ + tok) * HD + hc;
        const size_t idx_v  = ((size_t)(bb * NH + head) * HD + hc) * SEQ + tok;
        if (which == 0)      q[idx_qk] = v;
        else if (which == 1) kk[idx_qk] = v;
        else                 vt[idx_v] = v;   // V stored transposed (hd, seq)
      }
    }
  }
}

// ---------------------------------------------------------------------------
// Kernel 2: flash attention. grid = B*NH*(SEQ/128) = 768, block = 128 (4 waves).
// One wave owns 32 query rows (two 16-row tiles sharing K fragments); loops
// over 32-key tiles with online softmax.
// ---------------------------------------------------------------------------
__global__ __launch_bounds__(128) void attn_kernel(
    const __bf16* __restrict__ q, const __bf16* __restrict__ kmat,
    const __bf16* __restrict__ vt, __bf16* __restrict__ o) {
  __shared__ float lds[4][16 * 32];     // 2KB per wave: P re-layout C->A fragment
  const int wv = threadIdx.x >> 5;
  float* sp = lds[wv];

  const int bh   = blockIdx.x >> 5;     // (batch, head); 32 q-blocks of 128 rows
  const int qblk = blockIdx.x & 31;
  const int b = bh / NH, head = bh % NH;
  const int q0 = qblk * 128 + wv * 32;

  const __bf16* qh = q    + (size_t)bh * SEQ * HD;
  const __bf16* kh = kmat + (size_t)bh * SEQ * HD;
  const __bf16* vh = vt   + (size_t)bh * HD * SEQ;

  const int l = lane_id();
  const int cn = l & 15, hi = l >> 4;

  // Two Q tiles resident in registers: 2 x (16x64 bf16) = 4 A-fragments
  v16bf qa[2][2];
#pragma unroll
  for (int t = 0; t < 2; ++t) {
    qa[t][0] = frag_a_bf16(qh, HD, q0 + 16 * t, 0);
    qa[t][1] = frag_a_bf16(qh, HD, q0 + 16 * t, 32);
  }

  v8f oacc[2][4];
  float mi[2][8], li[2][8];
#pragma unroll
  for (int t = 0; t < 2; ++t) {
#pragma unroll
    for (int g = 0; g < 4; ++g) oacc[t][g] = vzero();
#pragma unroll
    for (int r = 0; r < 8; ++r) { mi[t][r] = -3.0e38f; li[t][r] = 0.0f; }
  }

  for (int kt = 0; kt < SEQ; kt += 32) {
    // K fragments loaded once, shared by both q-tiles
    v16bf kb[4];
    kb[0] = frag_b_bf16(kh, HD, kt,      0);
    kb[1] = frag_b_bf16(kh, HD, kt,      32);
    kb[2] = frag_b_bf16(kh, HD, kt + 16, 0);
    kb[3] = frag_b_bf16(kh, HD, kt + 16, 32);

    // S = (Q*scale) . K^T : per q-tile two 16x16 f32 C-fragments
    v8f s[2][2];
#pragma unroll
    for (int t = 0; t < 2; ++t) {
      s[t][0] = wmma_bf16(qa[t][0], kb[0], vzero());
      s[t][0] = wmma_bf16(qa[t][1], kb[1], s[t][0]);
      s[t][1] = wmma_bf16(qa[t][0], kb[2], vzero());
      s[t][1] = wmma_bf16(qa[t][1], kb[3], s[t][1]);
    }

#pragma unroll
    for (int t = 0; t < 2; ++t) {
      // Online softmax: C-fragment row m = 8*hi + r lives across a 16-lane half
#pragma unroll
      for (int r = 0; r < 8; ++r) {
        float tm = fmaxf(s[t][0][r], s[t][1][r]);
#pragma unroll
        for (int msk = 1; msk < 16; msk <<= 1) tm = fmaxf(tm, __shfl_xor(tm, msk, 16));
        const float nm = fmaxf(mi[t][r], tm);
        const float al = __expf(mi[t][r] - nm);
        mi[t][r] = nm;
        const float p0 = __expf(s[t][0][r] - nm);
        const float p1 = __expf(s[t][1][r] - nm);
        sp[(hi * 8 + r) * 32 + cn]      = p0;
        sp[(hi * 8 + r) * 32 + 16 + cn] = p1;
        float rs = p0 + p1;
#pragma unroll
        for (int msk = 1; msk < 16; msk <<= 1) rs += __shfl_xor(rs, msk, 16);
        li[t][r] = li[t][r] * al + rs;
#pragma unroll
        for (int g = 0; g < 4; ++g) oacc[t][g][r] *= al;
      }
      asm volatile("s_wait_dscnt 0x0" ::: "memory");  // same-wave LDS store->load

      // Rebuild P as 16x32 A-fragment (bf16) from LDS
      v16bf pf;
      {
        const float* row = sp + cn * 32;
        f32x4 x0 = *(const f32x4*)(row + hi * 8);
        f32x4 x1 = *(const f32x4*)(row + hi * 8 + 4);
        f32x4 x2 = *(const f32x4*)(row + 16 + hi * 8);
        f32x4 x3 = *(const f32x4*)(row + 16 + hi * 8 + 4);
#pragma unroll
        for (int i = 0; i < 4; ++i) {
          pf[i]      = (__bf16)x0[i];
          pf[4 + i]  = (__bf16)x1[i];
          pf[8 + i]  = (__bf16)x2[i];
          pf[12 + i] = (__bf16)x3[i];
        }
      }
      // O += P.V  (V^T layout: each lane holds keys kt+8*hi+0..7 and kt+16+8*hi+0..7)
      // LDS slab reuse across t is safe: same-wave LDS ops execute in order.
#pragma unroll
      for (int g = 0; g < 4; ++g)
        oacc[t][g] = wmma_bf16(pf, frag_b_bf16(vh, SEQ, g * 16, kt), oacc[t][g]);
    }
  }

  // Normalize and write (B, N, C) bf16 for the projection GEMM
#pragma unroll
  for (int t = 0; t < 2; ++t) {
#pragma unroll
    for (int r = 0; r < 8; ++r) {
      const float inv = 1.0f / li[t][r];
      const int tok = q0 + t * 16 + hi * 8 + r;
      const size_t base = ((size_t)b * SEQ + tok) * DIM + head * HD;
#pragma unroll
      for (int g = 0; g < 4; ++g)
        o[base + g * 16 + cn] = (__bf16)(oacc[t][g][r] * inv);
    }
  }
}

// ---------------------------------------------------------------------------
// Kernel 3: projection GEMM. grid=(8192/128, 768/256), block=256.
// ---------------------------------------------------------------------------
__global__ __launch_bounds__(256) void proj_kernel(
    const __bf16* __restrict__ a, const float* __restrict__ w,
    const float* __restrict__ bias, float* __restrict__ out) {
  const int wv = threadIdx.x >> 5;
  const int m0 = blockIdx.x * 128 + (wv >> 2) * 64;
  const int n0 = blockIdx.y * 256 + (wv & 3) * 64;
  const int l = lane_id();
  const int cn = l & 15, hi = l >> 4;

  v8f acc[4][4];
#pragma unroll
  for (int i = 0; i < 4; ++i)
#pragma unroll
    for (int j = 0; j < 4; ++j) acc[i][j] = vzero();

  for (int k0 = 0; k0 < DIM; k0 += 32) {
    if (k0 + 32 < DIM) {
      __builtin_prefetch(a + (size_t)(m0 + cn) * DIM + k0 + 32, 0, 1);
      __builtin_prefetch(w + (size_t)(n0 + cn) * DIM + k0 + 32, 0, 1);
    }
    v16bf av[4], bv[4];
#pragma unroll
    for (int i = 0; i < 4; ++i) av[i] = frag_a_bf16(a, DIM, m0 + 16 * i, k0);
#pragma unroll
    for (int j = 0; j < 4; ++j) bv[j] = frag_b_f32(w, DIM, n0 + 16 * j, k0);
#pragma unroll
    for (int i = 0; i < 4; ++i)
#pragma unroll
      for (int j = 0; j < 4; ++j) acc[i][j] = wmma_bf16(av[i], bv[j], acc[i][j]);
  }

#pragma unroll
  for (int j = 0; j < 4; ++j) {
    const int d = n0 + j * 16 + cn;
    const float bv = bias[d];
#pragma unroll
    for (int i = 0; i < 4; ++i) {
#pragma unroll
      for (int r = 0; r < 8; ++r) {
        const int m = m0 + i * 16 + hi * 8 + r;
        out[(size_t)m * DIM + d] = acc[i][j][r] + bv;
      }
    }
  }
}

// ---------------------------------------------------------------------------
extern "C" void kernel_launch(void* const* d_in, const int* in_sizes, int n_in,
                              void* d_out, int out_size, void* d_ws, size_t ws_size,
                              hipStream_t stream) {
  (void)in_sizes; (void)n_in; (void)out_size; (void)ws_size;
  const float* x      = (const float*)d_in[0];
  const float* qkv_w  = (const float*)d_in[1];
  const float* qkv_b  = (const float*)d_in[2];
  const float* proj_w = (const float*)d_in[3];
  const float* proj_b = (const float*)d_in[4];
  float* out = (float*)d_out;

  // Workspace: Q, K, V^T, attn-out (bf16) — ~50 MB total
  __bf16* q  = (__bf16*)d_ws;
  __bf16* k  = q  + QKV_ELEMS;
  __bf16* vt = k  + QKV_ELEMS;
  __bf16* o  = vt + QKV_ELEMS;

  qkv_kernel<<<dim3(8192 / 128, (3 * DIM) / 256), 256, 0, stream>>>(
      x, qkv_w, qkv_b, q, k, vt);
  attn_kernel<<<BATCH * NH * (SEQ / 128), 128, 0, stream>>>(q, k, vt, o);
  proj_kernel<<<dim3(8192 / 128, DIM / 256), 256, 0, stream>>>(
      o, proj_w, proj_b, out);
}
